// MyLSTMCell_22119081574758
// MI455X (gfx1250) — matrix operand order repaired and checked by reading, not compiled, not run on hardware
//
#include <hip/hip_runtime.h>

#define BDIM  4096
#define INDIM 2048
#define HDIM  2048
#define KDIM  (INDIM + HDIM)   // 4096

#define MT   128               // macro tile M (batch rows)      -> gridDim.y = 32
#define NT   64                // macro tile N (hidden cols)     -> gridDim.x = 32
#define KT   32                // K per WMMA step (bf16)
#define LDA  40                // padded LDS row stride in halves (32 + 8)
#define NKT  (KDIM / KT)       // 128 K-steps

typedef __attribute__((ext_vector_type(16))) __bf16 v16bf;
typedef __attribute__((ext_vector_type(8)))  __bf16 v8bf;
typedef __attribute__((ext_vector_type(8)))  float  v8f;

#if __has_builtin(__builtin_amdgcn_global_load_async_to_lds_b128) && \
    __has_builtin(__builtin_amdgcn_s_wait_asynccnt)
#define HAVE_ASYNC_LDS 1
typedef int v4i_t __attribute__((vector_size(16)));
typedef v4i_t __attribute__((address_space(1)))* gv4i_p;   // global int4*
typedef v4i_t __attribute__((address_space(3)))* lv4i_p;   // LDS int4*
#else
#define HAVE_ASYNC_LDS 0
#endif

// f32 -> packed 2x bf16 (round to nearest, ties away) in 3 VALU ops:
// two adds + one v_perm_b32 byte-permute grabbing the high halves.
// dst = { bf16(a) in [15:0], bf16(b) in [31:16] }
__device__ __forceinline__ unsigned int pack2(float a, float b) {
  unsigned int ua = __float_as_uint(a) + 0x8000u;
  unsigned int ub = __float_as_uint(b) + 0x8000u;
  // sel 0x07060302: dst.b0=S1.b2, dst.b1=S1.b3, dst.b2=S0.b2, dst.b3=S0.b3
  return __builtin_amdgcn_perm(ub, ua, 0x07060302u);
}

__device__ __forceinline__ float sigmoidf_(float x) {
  return 1.0f / (1.0f + __expf(-x));
}

// One-pass f32 -> bf16 conversion of the four weight matrices into d_ws.
// grid = (HDIM*KDIM/4/256, 4), block = 256. Each thread: float4 -> uint2.
__global__ __launch_bounds__(256)
void convert_weights(const float* __restrict__ Wf, const float* __restrict__ Wi,
                     const float* __restrict__ Wc, const float* __restrict__ Wo,
                     unsigned short* __restrict__ dst) {
  const float* Wg[4] = { Wf, Wi, Wc, Wo };
  const int g = blockIdx.y;
  const size_t idx = (size_t)blockIdx.x * 256 + threadIdx.x;   // float4 index
  const float4 v = *(const float4*)(Wg[g] + idx * 4);
  uint2* d = (uint2*)(dst + (size_t)g * HDIM * KDIM);
  d[idx] = (uint2){ pack2(v.x, v.y), pack2(v.z, v.w) };
}

template <bool PREBF16>
__global__ __launch_bounds__(256, 2)
void lstm_gates_wmma(const float* __restrict__ xt,
                     const float* __restrict__ ht,
                     const float* __restrict__ Wf,
                     const float* __restrict__ Wi,
                     const float* __restrict__ Wc,
                     const float* __restrict__ Wo,
                     const unsigned short* __restrict__ wsB,  // [4][HDIM][KDIM] bf16
                     const float* __restrict__ bF,
                     const float* __restrict__ bI,
                     const float* __restrict__ bC,
                     const float* __restrict__ bO,
                     float* __restrict__ out) {
  __shared__ unsigned short sA[MT * LDA];          // 10,240 B
  __shared__ unsigned short sB[4][NT * LDA];       // 20,480 B

  const int tid  = threadIdx.x;                    // 256 threads = 8 wave32
  const int lane = tid & 31;
  const int w    = tid >> 5;
  const int wm   = w >> 1;                         // 0..3  (M wave)
  const int wn   = w & 1;                          // 0..1  (N wave)
  const int lr   = lane & 15;
  const int hi   = lane >> 4;                      // lane half

  const int nBase = blockIdx.x * NT;
  const int mBase = blockIdx.y * MT;

  const float* Wg[4] = { Wf, Wi, Wc, Wo };

  v8f acc[4][2][2];
  #pragma unroll
  for (int g = 0; g < 4; ++g)
    #pragma unroll
    for (int mi = 0; mi < 2; ++mi)
      #pragma unroll
      for (int ni = 0; ni < 2; ++ni)
        acc[g][mi][ni] = (v8f){0.f,0.f,0.f,0.f,0.f,0.f,0.f,0.f};

  for (int kt = 0; kt < NKT; ++kt) {
    const int k0 = kt * KT;

    // ---------------- stage A tile (f32 -> bf16 into LDS) ----------------
    const float* srcA = (k0 < INDIM) ? xt : ht;
    const int    kkA  = (k0 < INDIM) ? k0 : (k0 - INDIM);
    #pragma unroll
    for (int j = 0; j < 4; ++j) {
      const int f   = tid + 256 * j;               // 1024 float4 chunks
      const int row = f >> 3;                      // 0..127
      const int kq  = (f & 7) * 4;
      const float4 v = *(const float4*)(srcA + (size_t)(mBase + row) * INDIM + kkA + kq);
      *(uint2*)&sA[row * LDA + kq] = (uint2){ pack2(v.x, v.y), pack2(v.z, v.w) };
    }
    // ---------------- stage 4 B tiles ----------------
    if (PREBF16) {
#if HAVE_ASYNC_LDS
      // bf16 weights: DMA straight into LDS, no VGPR round-trip (ASYNCcnt).
      // One b128 per lane per gate moves the whole 64x32-half gate tile.
      {
        const int f   = tid;
        const int row = f >> 2;                    // 0..63
        const int kq  = (f & 3) * 8;
        #pragma unroll
        for (int g = 0; g < 4; ++g) {
          const unsigned short* gp = wsB + (size_t)g * HDIM * KDIM +
                                     (size_t)(nBase + row) * KDIM + k0 + kq;
          __builtin_amdgcn_global_load_async_to_lds_b128(
              (gv4i_p)gp, (lv4i_p)&sB[g][row * LDA + kq], 0, 0);
        }
      }
      __builtin_amdgcn_s_wait_asynccnt(0);
#else
      #pragma unroll
      for (int g = 0; g < 4; ++g) {
        const unsigned short* Wp = wsB + (size_t)g * HDIM * KDIM;
        const int f   = tid;                       // 256 chunks of 8 halves
        const int row = f >> 2;                    // 0..63
        const int kq  = (f & 3) * 8;
        const uint4 v = *(const uint4*)(Wp + (size_t)(nBase + row) * KDIM + k0 + kq);
        *(uint4*)&sB[g][row * LDA + kq] = v;
      }
#endif
    } else {
      #pragma unroll
      for (int g = 0; g < 4; ++g) {
        const float* Wp = Wg[g];
        #pragma unroll
        for (int j = 0; j < 2; ++j) {
          const int f   = tid + 256 * j;           // 512 float4 chunks
          const int row = f >> 3;                  // 0..63
          const int kq  = (f & 7) * 4;
          const float4 v = *(const float4*)(Wp + (size_t)(nBase + row) * KDIM + k0 + kq);
          *(uint2*)&sB[g][row * LDA + kq] = (uint2){ pack2(v.x, v.y), pack2(v.z, v.w) };
        }
      }
    }
    __syncthreads();

    // prefetch next K tile into cache while WMMAs run (global_prefetch_b8)
    if (kt + 1 < NKT) {
      const int k1 = k0 + KT;
      const float* pA  = (k1 < INDIM) ? xt : ht;
      const int    k1a = (k1 < INDIM) ? k1 : (k1 - INDIM);
      {
        const int row = tid >> 3, kq = (tid & 7) * 4;
        __builtin_prefetch(pA + (size_t)(mBase + row) * INDIM + k1a + kq, 0, 1);
      }
      if (PREBF16) {
        const int row = tid >> 2, kq = (tid & 3) * 8;
        #pragma unroll
        for (int g = 0; g < 4; ++g)
          __builtin_prefetch(wsB + (size_t)g * HDIM * KDIM +
                             (size_t)(nBase + row) * KDIM + k1 + kq, 0, 1);
      } else {
        const int row = tid >> 3, kq = (tid & 7) * 4;
        #pragma unroll
        for (int g = 0; g < 4; ++g)
          __builtin_prefetch(Wg[g] + (size_t)(nBase + row) * KDIM + k1 + kq, 0, 1);
      }
    }

    // ---------------- compute: 16 WMMAs per wave per K-step ----------------
    union V16 { v16bf v; v8bf h[2]; };
    // A 16x32 bf16 layout: lane half 0 -> K 0-7 & 16-23, half 1 -> K 8-15 & 24-31
    V16 afr[2];
    #pragma unroll
    for (int mi = 0; mi < 2; ++mi) {
      const unsigned short* p = &sA[(wm * 32 + mi * 16 + lr) * LDA + hi * 8];
      afr[mi].h[0] = *(const v8bf*)(p);
      afr[mi].h[1] = *(const v8bf*)(p + 16);
    }
    #pragma unroll
    for (int g = 0; g < 4; ++g) {
      // Preload both N-subtile B fragments: 4 ds_load_b128 per wait,
      // then 4 WMMAs back-to-back.
      // B 32x16 bf16 layout: lane = N col, half 0 -> K 0-7 & 16-23, half 1 -> K 8-15 & 24-31
      V16 bfr[2];
      #pragma unroll
      for (int ni = 0; ni < 2; ++ni) {
        const unsigned short* p = &sB[g][(wn * 32 + ni * 16 + lr) * LDA + hi * 8];
        bfr[ni].h[0] = *(const v8bf*)(p);
        bfr[ni].h[1] = *(const v8bf*)(p + 16);
      }
      #pragma unroll
      for (int ni = 0; ni < 2; ++ni)
        #pragma unroll
        for (int mi = 0; mi < 2; ++mi)
          acc[g][mi][ni] = __builtin_amdgcn_wmma_f32_16x16x32_bf16(
              false, afr[mi].v, false, bfr[ni].v, (short)0, acc[g][mi][ni], false, false);
    }
    __syncthreads();
  }

  // ---------------- fused epilogue ----------------
  float* __restrict__ outH = out;
  float* __restrict__ outC = out + (size_t)BDIM * HDIM;
  #pragma unroll
  for (int mi = 0; mi < 2; ++mi) {
    #pragma unroll
    for (int ni = 0; ni < 2; ++ni) {
      const int gn = nBase + wn * 32 + ni * 16 + lr;
      const float vbF = bF[gn], vbI = bI[gn], vbC = bC[gn], vbO = bO[gn];
      #pragma unroll
      for (int i = 0; i < 8; ++i) {
        // C layout: VGPR i -> row i (lanes 0-15) / row i+8 (lanes 16-31)
        const int gm = mBase + wm * 32 + mi * 16 + i + hi * 8;
        const float F  = sigmoidf_(acc[0][mi][ni][i] + vbF);
        const float I  = sigmoidf_(acc[1][mi][ni][i] + vbI);
        const float Cv = tanhf   (acc[2][mi][ni][i] + vbC);
        const float O  = sigmoidf_(acc[3][mi][ni][i] + vbO);
        const float ct = (F + I) * Cv;               // faithful to source bug
        const float hv = O * tanhf(ct);
        outH[(size_t)gm * HDIM + gn] = hv;
        outC[(size_t)gm * HDIM + gn] = ct;
      }
    }
  }
}

extern "C" void kernel_launch(void* const* d_in, const int* in_sizes, int n_in,
                              void* d_out, int out_size, void* d_ws, size_t ws_size,
                              hipStream_t stream) {
  (void)in_sizes; (void)n_in; (void)out_size;
  const float* xt = (const float*)d_in[0];
  const float* ht = (const float*)d_in[1];
  // d_in[2] = ct_pro: unused (reference bug preserved)
  const float* Wf = (const float*)d_in[3];
  const float* bf = (const float*)d_in[4];
  const float* Wi = (const float*)d_in[5];
  const float* bi = (const float*)d_in[6];
  const float* Wc = (const float*)d_in[7];
  const float* bc = (const float*)d_in[8];
  const float* Wo = (const float*)d_in[9];
  const float* bo = (const float*)d_in[10];
  float* out = (float*)d_out;

  dim3 grid(HDIM / NT, BDIM / MT);   // (32, 32)
  dim3 block(256);

  const size_t needW = (size_t)4 * HDIM * KDIM * sizeof(unsigned short);  // 64 MB
  if (ws_size >= needW) {
    unsigned short* wsB = (unsigned short*)d_ws;
    dim3 cgrid((HDIM * (size_t)KDIM / 4) / 256, 4);   // (8192, 4)
    convert_weights<<<cgrid, block, 0, stream>>>(Wf, Wi, Wc, Wo, wsB);
    lstm_gates_wmma<true><<<grid, block, 0, stream>>>(xt, ht, Wf, Wi, Wc, Wo, wsB,
                                                      bf, bi, bc, bo, out);
  } else {
    lstm_gates_wmma<false><<<grid, block, 0, stream>>>(xt, ht, Wf, Wi, Wc, Wo, nullptr,
                                                       bf, bi, bc, bo, out);
  }
}
